// TriangleMultiplicationIncoming_31756988186829
// MI455X (gfx1250) — compile-verified, matrix operand order checked by reading, not run on hardware
//
#include <hip/hip_runtime.h>
#include <hip/hip_bf16.h>

// ---------------------------------------------------------------------------
// TriangleMultiplicationIncoming, MI455X (gfx1250, wave32, WMMA bf16)
//   z:(1,512,512,128) -> LN -> a,b,g = sigmoid(z@W.T)
//   p[i,j,c] = sum_k a[i,k,c]*b[k,j,c]   (128 channel-wise 512^3 GEMMs)
//   out = g * (LN(p) @ wf.T)
// Memory-bound (~70 GFLOP vs ~0.7 GB HBM traffic at 23.3 TB/s ≈ 30us floor);
// bf16 intermediates halve traffic, channel-major operand layouts make the
// einsum K-loop L2-resident (aT+bT2 = 134MB < 192MB L2). All WMMA fragments
// load as aligned 128-bit chunks (global_load_b128 / ds_load_b128).
// ---------------------------------------------------------------------------

typedef __bf16 v16bf __attribute__((ext_vector_type(16)));
typedef float  v8f   __attribute__((ext_vector_type(8)));
typedef unsigned int v8u   __attribute__((ext_vector_type(8)));
typedef unsigned int u32x4 __attribute__((ext_vector_type(4)));

#define SDIM 512
#define SS   (SDIM * SDIM)
#define CDIM 128
#define LNEPS 1e-5f

// ---- explicit global-address-space accessors (force global_load/store) ----
__device__ __forceinline__ u32x4 ldg_u32x4(const void* p) {
    return *(const __attribute__((address_space(1))) u32x4*)(unsigned long long)p;
}
__device__ __forceinline__ float ldg_f32(const void* p) {
    return *(const __attribute__((address_space(1))) float*)(unsigned long long)p;
}
__device__ __forceinline__ unsigned short ldg_u16(const void* p) {
    return *(const __attribute__((address_space(1))) unsigned short*)(unsigned long long)p;
}
__device__ __forceinline__ void stg_u16(void* p, unsigned short v) {
    *(__attribute__((address_space(1))) unsigned short*)(unsigned long long)p = v;
}
__device__ __forceinline__ void stg_f32(void* p, float v) {
    *(__attribute__((address_space(1))) float*)(unsigned long long)p = v;
}
__device__ __forceinline__ void stg_u32x4(void* p, u32x4 v) {
    *(__attribute__((address_space(1))) u32x4*)(unsigned long long)p = v;
}

__device__ __forceinline__ unsigned short f32_to_bf16_rte(float f) {
    unsigned int u = __builtin_bit_cast(unsigned int, f);
    u += 0x7FFFu + ((u >> 16) & 1u);   // round to nearest even
    return (unsigned short)(u >> 16);
}
__device__ __forceinline__ float bf16_bits_to_f32(unsigned short h) {
    unsigned int u = ((unsigned int)h) << 16;
    return __builtin_bit_cast(float, u);
}
// sigmoid with hardware v_exp_f32 + v_rcp_f32 (output is bf16 anyway)
__device__ __forceinline__ float sigmoidf_fast(float x) {
    return __builtin_amdgcn_rcpf(1.0f + __expf(-x));
}

// ---- WMMA 16-bit fragment loads (ISA 7.12.2) ------------------------------
// For lane (m = lane%16, half = lane/16) the A(16x32)/B(32x16) fragment is
// two contiguous 16B chunks of the k-fastest row:  [half*8, +8) and
// [16+half*8, +8) bf16 elements.  Row base is 16B aligned by construction.
__device__ __forceinline__ v16bf load_frag_g(const unsigned short* row_kb, int half) {
    u32x4 lo = ldg_u32x4(row_kb + half * 8);        // global_load_b128
    u32x4 hi = ldg_u32x4(row_kb + 16 + half * 8);   // global_load_b128
    v8u u = __builtin_shufflevector(lo, hi, 0, 1, 2, 3, 4, 5, 6, 7);
    return __builtin_bit_cast(v16bf, u);
}
__device__ __forceinline__ v16bf load_frag_lds(const unsigned short* row_kb, int half) {
    u32x4 lo = *(const u32x4*)(row_kb + half * 8);        // ds_load_b128
    u32x4 hi = *(const u32x4*)(row_kb + 16 + half * 8);   // ds_load_b128
    v8u u = __builtin_shufflevector(lo, hi, 0, 1, 2, 3, 4, 5, 6, 7);
    return __builtin_bit_cast(v16bf, u);
}

// ---------------------------------------------------------------------------
// Kernel 0: fp32 -> bf16 weight cast
// ---------------------------------------------------------------------------
__global__ void cvt_w_kernel(const float* __restrict__ w,
                             unsigned short* __restrict__ o, int n) {
    int i = blockIdx.x * blockDim.x + threadIdx.x;
    if (i < n) stg_u16(o + i, f32_to_bf16_rte(ldg_f32(w + i)));
}

// ---------------------------------------------------------------------------
// Kernel 1: fused LayerNorm + three 128x128 projections + sigmoid.
// One block = 16 rows of z. Outputs:
//   aT  [c][i][k]  (k fastest)   bT2 [c][j][k] (k fastest)   g [r][c]
// ---------------------------------------------------------------------------
__global__ __launch_bounds__(128)
void ln_proj_kernel(const float* __restrict__ z,
                    const float* __restrict__ ln_w,
                    const float* __restrict__ ln_b,
                    const unsigned short* __restrict__ wa,
                    const unsigned short* __restrict__ wb,
                    const unsigned short* __restrict__ wg,
                    unsigned short* __restrict__ aT,
                    unsigned short* __restrict__ bT2,
                    unsigned short* __restrict__ gmat) {
    __shared__ alignas(16) unsigned short zn[16 * CDIM];   // normalized rows, bf16
    __shared__ float redsum[128], redsq[128];
    __shared__ float smu[16], srs[16];

    const int t  = threadIdx.x;
    const int r0 = blockIdx.x * 16;                   // 16 consecutive rows

    // ---- LayerNorm: thread t -> row t/8, 16-col chunk (t%8) ----
    const int row = t >> 3;
    const int q   = t & 7;
    float vbuf[16];
    float s = 0.f, s2 = 0.f;
    const float* zr = z + (size_t)(r0 + row) * CDIM + q * 16;
    #pragma unroll
    for (int e = 0; e < 16; ++e) { float x = ldg_f32(zr + e); vbuf[e] = x; s += x; s2 += x * x; }
    redsum[t] = s; redsq[t] = s2;
    __syncthreads();
    if (q == 0) {
        float ts = 0.f, t2 = 0.f;
        #pragma unroll
        for (int e = 0; e < 8; ++e) { ts += redsum[row * 8 + e]; t2 += redsq[row * 8 + e]; }
        float mu  = ts * (1.0f / CDIM);
        float var = t2 * (1.0f / CDIM) - mu * mu;
        smu[row] = mu; srs[row] = __builtin_amdgcn_rsqf(var + LNEPS);
    }
    __syncthreads();
    {
        float mu = smu[row], rs = srs[row];
        #pragma unroll
        for (int e = 0; e < 16; ++e) {
            int col = q * 16 + e;
            float nx = (vbuf[e] - mu) * rs * ldg_f32(ln_w + col) + ldg_f32(ln_b + col);
            zn[row * CDIM + col] = f32_to_bf16_rte(nx);
        }
    }
    __syncthreads();

    // ---- 16x128x128 GEMMs: wave w owns output cols [w*32, w*32+32) ----
    const int lane = t & 31;
    const int wave = t >> 5;
    const int m    = lane & 15;       // A: row M, B: col N, D: col N
    const int half = lane >> 4;

    // A fragments are loop-invariant across the three weight matrices:
    // hoist the four K-step fragments out of the wsel loop.
    v16bf afr[4];
    #pragma unroll
    for (int ks = 0; ks < 4; ++ks)
        afr[ks] = load_frag_lds(&zn[m * CDIM + ks * 32], half);

    const unsigned short* Ws[3] = { wa, wb, wg };
    #pragma unroll 1
    for (int wsel = 0; wsel < 3; ++wsel) {
        const unsigned short* W = Ws[wsel];
        const unsigned short* Wr0 = W + (wave * 32 + m) * CDIM;        // col n0 row
        const unsigned short* Wr1 = Wr0 + 16 * CDIM;                   // col n1 row
        v8f acc0 = {}; v8f acc1 = {};
        #pragma unroll
        for (int ks = 0; ks < 4; ++ks) {
            const int kb = ks * 32;
            v16bf bf0 = load_frag_g(Wr0 + kb, half);
            v16bf bf1 = load_frag_g(Wr1 + kb, half);
            acc0 = __builtin_amdgcn_wmma_f32_16x16x32_bf16(false, afr[ks], false, bf0,
                                                           (short)0, acc0, false, false);
            acc1 = __builtin_amdgcn_wmma_f32_16x16x32_bf16(false, afr[ks], false, bf1,
                                                           (short)0, acc1, false, false);
        }
        // D layout: lane -> N = lane%16, rows = half*8 + vr  (vr = VGPR idx)
        #pragma unroll
        for (int tile = 0; tile < 2; ++tile) {
            v8f acc = tile ? acc1 : acc0;
            const int c = wave * 32 + tile * 16 + m;
            if (wsel == 0) {
                // a -> aT[c][i][k] : rows contiguous -> one 16B store/lane
                u32x4 pk;
                #pragma unroll
                for (int vv = 0; vv < 4; ++vv) {
                    unsigned short lo = f32_to_bf16_rte(sigmoidf_fast(acc[2 * vv]));
                    unsigned short hi = f32_to_bf16_rte(sigmoidf_fast(acc[2 * vv + 1]));
                    pk[vv] = (unsigned int)lo | ((unsigned int)hi << 16);
                }
                stg_u32x4(&aT[(size_t)c * SS + r0 + half * 8], pk);
            } else if (wsel == 1) {
                // b -> bT2[c][j][k]: row r = x*S + y; scatter with stride S
                const int x = r0 >> 9, y0 = r0 & (SDIM - 1);
                #pragma unroll
                for (int vr = 0; vr < 8; ++vr)
                    stg_u16(&bT2[(size_t)c * SS + (size_t)(y0 + half * 8 + vr) * SDIM + x],
                            f32_to_bf16_rte(sigmoidf_fast(acc[vr])));
            } else {
                // g -> natural (r, c) layout
                #pragma unroll
                for (int vr = 0; vr < 8; ++vr)
                    stg_u16(&gmat[(size_t)(r0 + half * 8 + vr) * CDIM + c],
                            f32_to_bf16_rte(sigmoidf_fast(acc[vr])));
            }
        }
    }
}

// ---------------------------------------------------------------------------
// Kernel 2: per-channel einsum  P_c = A_c @ B_c (B stored as [j][k]).
// One wave computes a 16x64 tile of one channel; A frag reused 4x.
// aT+bT2 = 134MB -> resident in 192MB L2 across the K loop.
// ---------------------------------------------------------------------------
__global__ __launch_bounds__(32)
void einsum_kernel(const unsigned short* __restrict__ aT,
                   const unsigned short* __restrict__ bT2,
                   float* __restrict__ pT) {
    const int lane = threadIdx.x;
    const int m    = lane & 15;
    const int half = lane >> 4;
    const int i0   = blockIdx.x * 16;
    const int j0   = blockIdx.y * 64;
    const int c    = blockIdx.z;

    const unsigned short* Ab = aT  + (size_t)c * SS + (size_t)(i0 + m) * SDIM;
    const unsigned short* B0 = bT2 + (size_t)c * SS + (size_t)(j0 + m) * SDIM;

    v8f acc[4] = { v8f{}, v8f{}, v8f{}, v8f{} };
    for (int kb = 0; kb < SDIM; kb += 32) {
        v16bf af = load_frag_g(Ab + kb, half);
        #pragma unroll
        for (int jt = 0; jt < 4; ++jt) {
            v16bf bf = load_frag_g(B0 + (size_t)jt * 16 * SDIM + kb, half);
            acc[jt] = __builtin_amdgcn_wmma_f32_16x16x32_bf16(false, af, false, bf,
                                                              (short)0, acc[jt], false, false);
        }
    }
    // store pT[c][i][j] (f32, keeps final-LN numerics)
    #pragma unroll
    for (int jt = 0; jt < 4; ++jt) {
        const int n = j0 + jt * 16 + m;
        float* pp = pT + (size_t)c * SS + (size_t)(i0 + half * 8) * SDIM + n;
        #pragma unroll
        for (int vr = 0; vr < 8; ++vr)
            stg_f32(pp + (size_t)vr * SDIM, acc[jt][vr]);
    }
}

// ---------------------------------------------------------------------------
// Kernel 3: fused final LayerNorm + wf projection + gate multiply.
// ---------------------------------------------------------------------------
__global__ __launch_bounds__(128)
void fln_out_kernel(const float* __restrict__ pT,
                    const float* __restrict__ fln_w,
                    const float* __restrict__ fln_b,
                    const unsigned short* __restrict__ wf,
                    const unsigned short* __restrict__ gmat,
                    float* __restrict__ out) {
    __shared__ alignas(16) float          pt[16 * CDIM];
    __shared__ alignas(16) unsigned short zn[16 * CDIM];
    __shared__ float redsum[128], redsq[128];
    __shared__ float smu[16], srs[16];

    const int t  = threadIdx.x;
    const int r0 = blockIdx.x * 16;

    // gather p rows from channel-major pT: 64B contiguous per (c, 16 rows)
    #pragma unroll 1
    for (int pass = 0; pass < 16; ++pass) {
        const int c = (t >> 4) + pass * 8;
        const int m = t & 15;
        pt[m * CDIM + c] = ldg_f32(&pT[(size_t)c * SS + r0 + m]);
    }
    __syncthreads();

    // LayerNorm over C
    const int row = t >> 3, q = t & 7;
    float s = 0.f, s2 = 0.f;
    #pragma unroll
    for (int e = 0; e < 16; ++e) {
        float x = pt[row * CDIM + q * 16 + e]; s += x; s2 += x * x;
    }
    redsum[t] = s; redsq[t] = s2;
    __syncthreads();
    if (q == 0) {
        float ts = 0.f, t2 = 0.f;
        #pragma unroll
        for (int e = 0; e < 8; ++e) { ts += redsum[row * 8 + e]; t2 += redsq[row * 8 + e]; }
        float mu  = ts * (1.0f / CDIM);
        float var = t2 * (1.0f / CDIM) - mu * mu;
        smu[row] = mu; srs[row] = __builtin_amdgcn_rsqf(var + LNEPS);
    }
    __syncthreads();
    {
        float mu = smu[row], rs = srs[row];
        #pragma unroll
        for (int e = 0; e < 16; ++e) {
            int col = q * 16 + e;
            float nx = (pt[row * CDIM + col] - mu) * rs * ldg_f32(fln_w + col)
                       + ldg_f32(fln_b + col);
            zn[row * CDIM + col] = f32_to_bf16_rte(nx);
        }
    }
    __syncthreads();

    // GEMM with wf, then gate with g
    const int lane = t & 31;
    const int wave = t >> 5;
    const int m    = lane & 15;
    const int half = lane >> 4;

    const unsigned short* Wr0 = wf + (wave * 32 + m) * CDIM;
    const unsigned short* Wr1 = Wr0 + 16 * CDIM;

    v8f acc0 = {}; v8f acc1 = {};
    #pragma unroll
    for (int ks = 0; ks < 4; ++ks) {
        const int kb = ks * 32;
        v16bf af  = load_frag_lds(&zn[m * CDIM + kb], half);
        v16bf bf0 = load_frag_g(Wr0 + kb, half);
        v16bf bf1 = load_frag_g(Wr1 + kb, half);
        acc0 = __builtin_amdgcn_wmma_f32_16x16x32_bf16(false, af, false, bf0,
                                                       (short)0, acc0, false, false);
        acc1 = __builtin_amdgcn_wmma_f32_16x16x32_bf16(false, af, false, bf1,
                                                       (short)0, acc1, false, false);
    }
    #pragma unroll
    for (int tile = 0; tile < 2; ++tile) {
        v8f acc = tile ? acc1 : acc0;
        const int c = wave * 32 + tile * 16 + m;
        #pragma unroll
        for (int vr = 0; vr < 8; ++vr) {
            const size_t ridx = (size_t)(r0 + half * 8 + vr) * CDIM + c;
            stg_f32(out + ridx, acc[vr] * bf16_bits_to_f32(ldg_u16(gmat + ridx)));
        }
    }
}

// ---------------------------------------------------------------------------
// Host launcher
// ---------------------------------------------------------------------------
extern "C" void kernel_launch(void* const* d_in, const int* in_sizes, int n_in,
                              void* d_out, int out_size, void* d_ws, size_t ws_size,
                              hipStream_t stream) {
    const float* z    = (const float*)d_in[0];
    const float* ln_w = (const float*)d_in[1];
    const float* ln_b = (const float*)d_in[2];
    const float* wa   = (const float*)d_in[3];
    const float* wb   = (const float*)d_in[4];
    const float* wg   = (const float*)d_in[5];
    const float* flw  = (const float*)d_in[6];
    const float* flb  = (const float*)d_in[7];
    const float* wf   = (const float*)d_in[8];
    float* out = (float*)d_out;

    // workspace carve-up (all offsets 16B-aligned)
    unsigned short* wa_h = (unsigned short*)d_ws;
    unsigned short* wb_h = wa_h + 16384;
    unsigned short* wg_h = wb_h + 16384;
    unsigned short* wf_h = wg_h + 16384;
    unsigned short* aT   = wf_h + 16384;                 // [C][S][S] bf16, 67MB
    unsigned short* bT2  = aT  + (size_t)SS * CDIM;      // [C][S][S] bf16, 67MB
    unsigned short* gmat = bT2 + (size_t)SS * CDIM;      // [S*S][C]  bf16, 67MB
    float*          pT   = (float*)(gmat + (size_t)SS * CDIM); // [C][S][S] f32, 134MB

    cvt_w_kernel<<<64, 256, 0, stream>>>(wa, wa_h, 16384);
    cvt_w_kernel<<<64, 256, 0, stream>>>(wb, wb_h, 16384);
    cvt_w_kernel<<<64, 256, 0, stream>>>(wg, wg_h, 16384);
    cvt_w_kernel<<<64, 256, 0, stream>>>(wf, wf_h, 16384);

    ln_proj_kernel<<<SS / 16, 128, 0, stream>>>(z, ln_w, ln_b, wa_h, wb_h, wg_h,
                                                aT, bT2, gmat);

    einsum_kernel<<<dim3(SDIM / 16, SDIM / 64, CDIM), 32, 0, stream>>>(aT, bT2, pT);

    fln_out_kernel<<<SS / 16, 128, 0, stream>>>(pT, flw, flb, wf_h, gmat, out);
}
